// SpikeTimingDependentPlasticity_34187939676381
// MI455X (gfx1250) — matrix operand order repaired and checked by reading, not compiled, not run on hardware
//
#include <hip/hip_runtime.h>

typedef __bf16 bf16_t;
typedef __attribute__((ext_vector_type(16))) __bf16 bf16x16;
typedef __attribute__((ext_vector_type(8)))  float  f32x8;

struct alignas(16) U4 { unsigned int x, y, z, w; };

#define S_DIM 2048
#define D_DIM 2048
#define B_DIM 4
#define M1    (B_DIM * S_DIM)     // 8192 rows of x
#define SD    (S_DIM * D_DIM)     // 4194304

#define A_PLUS  0.01f
#define A_MINUS 0.01f
#define DECAY   0.9512294245007140f   // exp(-1/20); TAU_PLUS == TAU_MINUS

#define BM 128
#define BN 128
#define BK 32
#define LDT 40                    // padded LDS row stride (elements)

// ---------------- small elementwise kernels ----------------

__global__ void __launch_bounds__(256) k_split(const float* __restrict__ src,
                                               bf16_t* __restrict__ hi,
                                               bf16_t* __restrict__ lo, int n) {
  int i = blockIdx.x * 256 + threadIdx.x;
  if (i >= n) return;
  float f = src[i];
  bf16_t h = (bf16_t)f;
  hi[i] = h;
  lo[i] = (bf16_t)(f - (float)h);
}

// split x into bf16 hi/lo and build P = mean_b(x > 0.5) (and its transpose)
__global__ void __launch_bounds__(256) k_splitx_p(const float* __restrict__ x,
                                                  bf16_t* __restrict__ xh,
                                                  bf16_t* __restrict__ xl,
                                                  bf16_t* __restrict__ P,
                                                  bf16_t* __restrict__ Pt) {
  int idx = blockIdx.x * 256 + threadIdx.x;    // (s,d) flat in [0, SD)
  int s = idx >> 11;
  int d = idx & (D_DIM - 1);
  float cnt = 0.f;
#pragma unroll
  for (int b = 0; b < B_DIM; ++b) {
    int o = b * SD + idx;
    float f = x[o];
    bf16_t h = (bf16_t)f;
    xh[o] = h;
    xl[o] = (bf16_t)(f - (float)h);
    if (f > 0.5f) cnt += 1.f;
  }
  bf16_t pm = (bf16_t)(cnt * 0.25f);           // exactly representable in bf16
  P[idx] = pm;
  Pt[(size_t)d * S_DIM + s] = pm;
}

// analytic STDP kernel matrix K[a][b], split into bf16 hi/lo
__global__ void __launch_bounds__(256) k_genK(bf16_t* __restrict__ Kh,
                                              bf16_t* __restrict__ Kl) {
  int idx = blockIdx.x * 256 + threadIdx.x;    // (a,b) flat in [0, S*S)
  int a = idx >> 11;
  int b = idx & (S_DIM - 1);
  int dlt = a - b;
  float kv = 0.f;
  if (dlt >= 0) kv += A_PLUS * expf(-(float)dlt * 0.05f);   // LTP: dm^(a-b)
  if (dlt <= 0) kv -= A_MINUS * expf((float)dlt * 0.05f);   // LTD: dp^(b-a)
  bf16_t h = (bf16_t)kv;
  Kh[idx] = h;
  Kl[idx] = (bf16_t)(kv - (float)h);
}

// Q = mean_b(out > 0.5)
__global__ void __launch_bounds__(256) k_q(const float* __restrict__ out,
                                           bf16_t* __restrict__ Q) {
  int idx = blockIdx.x * 256 + threadIdx.x;
  float cnt = 0.f;
#pragma unroll
  for (int b = 0; b < B_DIM; ++b)
    if (out[b * SD + idx] > 0.5f) cnt += 1.f;
  Q[idx] = (bf16_t)(cnt * 0.25f);
}

// u[i]=sum_s dm^{s+1} P[s,i];  v[j]=sum_s dp^{s+1} Q[s,j];
// new_pre[d]=dp^S*pre0[d]+sum_s dp^{S-1-s}P[s,d]; new_post likewise with Q.
__global__ void __launch_bounds__(256) k_traces(const bf16_t* __restrict__ P,
                                                const bf16_t* __restrict__ Q,
                                                const float* __restrict__ pre0,
                                                const float* __restrict__ post0,
                                                float* __restrict__ u,
                                                float* __restrict__ v,
                                                float* __restrict__ new_pre,
                                                float* __restrict__ new_post) {
  int d = blockIdx.x * 256 + threadIdx.x;      // [0, D)
  float au = 0.f, av = 0.f, w = DECAY;
  for (int s = 0; s < S_DIM; ++s) {
    au += w * (float)P[(size_t)s * D_DIM + d];
    av += w * (float)Q[(size_t)s * D_DIM + d];
    w *= DECAY;
  }
  u[d] = au;
  v[d] = av;
  float ap = 0.f, aq = 0.f, w2 = 1.f;
  for (int s = S_DIM - 1; s >= 0; --s) {
    ap += w2 * (float)P[(size_t)s * D_DIM + d];
    aq += w2 * (float)Q[(size_t)s * D_DIM + d];
    w2 *= DECAY;
  }
  new_pre[d]  = w2 * pre0[d]  + ap;            // w2 == DECAY^S
  new_post[d] = w2 * post0[d] + aq;
}

// ---------------- WMMA GEMM (bf16 split, f32 accumulate) ----------------
// Double-buffered; A tiles staged by GLOBAL_LOAD_ASYNC_TO_LDS_B128 (ASYNCcnt),
// B tiles staged transposed through registers -> ds_store.
// MODE 0: C = Ah*Bh + Ah*Bl + Al*Bh          -> f32 Cf          (out = x@W)
// MODE 1: C = Ah*Bh + Al*Bh                  -> bf16 Coh/Col    (T = K@Q)
// MODE 2: C = Ah*Bh + Ah*Bl; epilogue adds W and rank-1 terms   (new_W)
template <int MODE>
__global__ void __launch_bounds__(256) k_gemm(
    const bf16_t* __restrict__ Ah, const bf16_t* __restrict__ Al,
    const bf16_t* __restrict__ Bh, const bf16_t* __restrict__ Bl,
    float* __restrict__ Cf, bf16_t* __restrict__ Coh, bf16_t* __restrict__ Col,
    const float* __restrict__ Wold, const float* __restrict__ u,
    const float* __restrict__ v, const float* __restrict__ pre0,
    const float* __restrict__ post0, int M, int N, int K) {
  constexpr bool USE_AL = (MODE == 0 || MODE == 1);
  constexpr bool USE_BL = (MODE == 0 || MODE == 2);

  __shared__ bf16_t sAh[2][BM * LDT];
  __shared__ bf16_t sAl[2][USE_AL ? BM * LDT : 1];
  __shared__ bf16_t sBh[2][BN * LDT];               // stored transposed: [n][k]
  __shared__ bf16_t sBl[2][USE_BL ? BN * LDT : 1];

  const int tid   = threadIdx.x;
  const int lane  = tid & 31;
  const int wv    = tid >> 5;
  const int mr    = lane & 15;
  const int half  = lane >> 4;
  const int waveM = wv >> 1;                     // 0..3
  const int waveN = wv & 1;                      // 0..1
  const int row0  = blockIdx.y * BM;
  const int col0  = blockIdx.x * BN;

  f32x8 acc[2][4];
#pragma unroll
  for (int i = 0; i < 2; ++i)
#pragma unroll
    for (int j = 0; j < 4; ++j) acc[i][j] = f32x8{};

  const int ar = tid >> 1;                       // A stage: row 0..127
  const int ac = (tid & 1) << 4;                 // 0 or 16
  const int bk = tid >> 3;                       // B stage: k 0..31
  const int bn = (tid & 7) << 4;                 // n offset 0..112

  // Async global->LDS copy of this thread's 32-byte A chunk (2 x b128).
  // INST_OFFSET is added to BOTH the LDS and global address (ISA 08 §4.4),
  // so one vdst/vaddr pair + offset:16 covers both halves.
  auto issueA = [&](const bf16_t* __restrict__ G, bf16_t* sdst, int k0) {
    unsigned ldsa = (unsigned)(unsigned long long)(void*)(sdst + ar * LDT + ac);
    unsigned goff = (unsigned)(((size_t)(row0 + ar) * K + k0 + ac) * sizeof(bf16_t));
    asm volatile(
        "global_load_async_to_lds_b128 %0, %1, %2\n\t"
        "global_load_async_to_lds_b128 %0, %1, %2 offset:16"
        :
        : "v"(ldsa), "v"(goff), "s"(G)
        : "memory");
  };

  struct BReg { U4 q0, q1; };
  auto loadB = [&](const bf16_t* __restrict__ G, int k0) {
    const U4* g = reinterpret_cast<const U4*>(G + (size_t)(k0 + bk) * N + col0 + bn);
    BReg r;
    r.q0 = g[0];
    r.q1 = g[1];
    return r;
  };
  auto scatterB = [&](BReg r, bf16_t* sdst) {
    union { U4 q[2]; bf16_t e[16]; } t;
    t.q[0] = r.q0;
    t.q[1] = r.q1;
#pragma unroll
    for (int j = 0; j < 16; ++j) sdst[(bn + j) * LDT + bk] = t.e[j];
  };

  // ---- prologue: fill buffer 0 ----
  issueA(Ah, sAh[0], 0);
  if (USE_AL) issueA(Al, sAl[0], 0);
  BReg rBh = loadB(Bh, 0);
  BReg rBl{};
  if (USE_BL) rBl = loadB(Bl, 0);
  scatterB(rBh, sBh[0]);
  if (USE_BL) scatterB(rBl, sBl[0]);
  asm volatile("s_wait_asynccnt 0" ::: "memory");
  __syncthreads();

  const int NT = K / BK;
  for (int kt = 0; kt < NT; ++kt) {
    const int cur = kt & 1;
    const int nxt = cur ^ 1;
    const bool hasNext = (kt + 1) < NT;
    if (hasNext) {  // overlap next tile's global traffic with this tile's WMMA
      int kn = (kt + 1) * BK;
      issueA(Ah, sAh[nxt], kn);
      if (USE_AL) issueA(Al, sAl[nxt], kn);
      rBh = loadB(Bh, kn);
      if (USE_BL) rBl = loadB(Bl, kn);
    }

    const bf16_t* cAh = sAh[cur];
    const bf16_t* cAl = sAl[cur];
    const bf16_t* cBh = sBh[cur];
    const bf16_t* cBl = sBl[cur];

    union Frag { bf16x16 v; U4 q[2]; };
    Frag fAh[2], fAl[2], fBh[4], fBl[4];
#pragma unroll
    for (int mt = 0; mt < 2; ++mt) {
      // ISA A layout: lane m=lane%16; k-chunks [half*8, +8) and [16+half*8, +8)
      int ao = (waveM * 32 + mt * 16 + mr) * LDT + half * 8;
      fAh[mt].q[0] = *reinterpret_cast<const U4*>(cAh + ao);
      fAh[mt].q[1] = *reinterpret_cast<const U4*>(cAh + ao + 16);
      if (USE_AL) {
        fAl[mt].q[0] = *reinterpret_cast<const U4*>(cAl + ao);
        fAl[mt].q[1] = *reinterpret_cast<const U4*>(cAl + ao + 16);
      }
    }
#pragma unroll
    for (int nt = 0; nt < 4; ++nt) {
      // ISA B layout: lane n=lane%16; k-chunks [half*8, +8) and [16+half*8, +8)
      int bo = (waveN * 64 + nt * 16 + mr) * LDT + half * 8;
      fBh[nt].q[0] = *reinterpret_cast<const U4*>(cBh + bo);
      fBh[nt].q[1] = *reinterpret_cast<const U4*>(cBh + bo + 16);
      if (USE_BL) {
        fBl[nt].q[0] = *reinterpret_cast<const U4*>(cBl + bo);
        fBl[nt].q[1] = *reinterpret_cast<const U4*>(cBl + bo + 16);
      }
    }

#pragma unroll
    for (int mt = 0; mt < 2; ++mt)
#pragma unroll
      for (int nt = 0; nt < 4; ++nt) {
        acc[mt][nt] = __builtin_amdgcn_wmma_f32_16x16x32_bf16(
            false, fAh[mt].v, false, fBh[nt].v, (short)0, acc[mt][nt], false, false);
        if (USE_BL)
          acc[mt][nt] = __builtin_amdgcn_wmma_f32_16x16x32_bf16(
              false, fAh[mt].v, false, fBl[nt].v, (short)0, acc[mt][nt], false, false);
        if (USE_AL)
          acc[mt][nt] = __builtin_amdgcn_wmma_f32_16x16x32_bf16(
              false, fAl[mt].v, false, fBh[nt].v, (short)0, acc[mt][nt], false, false);
      }

    if (hasNext) {
      scatterB(rBh, sBh[nxt]);
      if (USE_BL) scatterB(rBl, sBl[nxt]);
      asm volatile("s_wait_asynccnt 0" ::: "memory");  // A(nxt) landed in LDS
    }
    __syncthreads();
  }

  // epilogue; C/D layout: VGPR e holds m = e + 8*half, n = lane%16
#pragma unroll
  for (int mt = 0; mt < 2; ++mt) {
    int grow = row0 + waveM * 32 + mt * 16 + half * 8;
#pragma unroll
    for (int nt = 0; nt < 4; ++nt) {
      int gcol = col0 + waveN * 64 + nt * 16 + mr;
      size_t base = (size_t)grow * N + gcol;
#pragma unroll
      for (int e = 0; e < 8; ++e) {
        size_t idx = base + (size_t)e * N;
        float t = acc[mt][nt][e];
        if constexpr (MODE == 0) {
          Cf[idx] = t;
        } else if constexpr (MODE == 1) {
          bf16_t th = (bf16_t)t;
          Coh[idx] = th;
          Col[idx] = (bf16_t)(t - (float)th);
        } else {
          Cf[idx] = Wold[idx] + t + A_PLUS * u[grow + e] * post0[gcol]
                    - A_MINUS * pre0[grow + e] * v[gcol];
        }
      }
    }
  }
}

// ---------------- launcher ----------------

extern "C" void kernel_launch(void* const* d_in, const int* in_sizes, int n_in,
                              void* d_out, int out_size, void* d_ws, size_t ws_size,
                              hipStream_t stream) {
  (void)in_sizes; (void)n_in; (void)out_size; (void)ws_size;

  const float* x     = (const float*)d_in[0];
  const float* W     = (const float*)d_in[1];
  const float* pre0  = (const float*)d_in[2];
  const float* post0 = (const float*)d_in[3];

  float* out      = (float*)d_out;                       // [B,S,D]
  float* newW     = out + (size_t)M1 * D_DIM;            // [D,D]
  float* new_pre  = newW + (size_t)D_DIM * D_DIM;        // [D]
  float* new_post = new_pre + D_DIM;                     // [D]

  char* p = (char*)d_ws;
  auto take = [&](size_t bytes) {
    char* r = p;
    p += (bytes + 255) & ~(size_t)255;
    return r;
  };
  bf16_t* xh = (bf16_t*)take((size_t)M1 * D_DIM * 2);
  bf16_t* xl = (bf16_t*)take((size_t)M1 * D_DIM * 2);
  bf16_t* Wh = (bf16_t*)take((size_t)D_DIM * D_DIM * 2);
  bf16_t* Wl = (bf16_t*)take((size_t)D_DIM * D_DIM * 2);
  bf16_t* P  = (bf16_t*)take((size_t)S_DIM * D_DIM * 2);
  bf16_t* Pt = (bf16_t*)take((size_t)D_DIM * S_DIM * 2);
  bf16_t* Q  = (bf16_t*)take((size_t)S_DIM * D_DIM * 2);
  bf16_t* Kh = (bf16_t*)take((size_t)S_DIM * S_DIM * 2);
  bf16_t* Kl = (bf16_t*)take((size_t)S_DIM * S_DIM * 2);
  bf16_t* Th = (bf16_t*)take((size_t)S_DIM * D_DIM * 2);
  bf16_t* Tl = (bf16_t*)take((size_t)S_DIM * D_DIM * 2);
  float*  u  = (float*)take(D_DIM * sizeof(float));
  float*  v  = (float*)take(D_DIM * sizeof(float));

  // 1) operand preparation
  k_split<<<(D_DIM * D_DIM) / 256, 256, 0, stream>>>(W, Wh, Wl, D_DIM * D_DIM);
  k_splitx_p<<<SD / 256, 256, 0, stream>>>(x, xh, xl, P, Pt);
  k_genK<<<(S_DIM * S_DIM) / 256, 256, 0, stream>>>(Kh, Kl);

  // 2) out = x @ W  (split-bf16, ~f32 accuracy)
  k_gemm<0><<<dim3(D_DIM / BN, M1 / BM), 256, 0, stream>>>(
      xh, xl, Wh, Wl, out, nullptr, nullptr,
      nullptr, nullptr, nullptr, nullptr, nullptr, M1, D_DIM, D_DIM);

  // 3) Q = mean_b(out > 0.5)
  k_q<<<SD / 256, 256, 0, stream>>>(out, Q);

  // 4) T = K @ Q (split K), split-stored as bf16 hi/lo
  k_gemm<1><<<dim3(D_DIM / BN, S_DIM / BM), 256, 0, stream>>>(
      Kh, Kl, Q, nullptr, nullptr, Th, Tl,
      nullptr, nullptr, nullptr, nullptr, nullptr, S_DIM, D_DIM, S_DIM);

  // 5) trace reductions (u, v, new_pre, new_post)
  k_traces<<<D_DIM / 256, 256, 0, stream>>>(P, Q, pre0, post0, u, v, new_pre, new_post);

  // 6) new_W = W + P^T @ (Th + Tl) + rank-1 initial-trace corrections
  k_gemm<2><<<dim3(D_DIM / BN, D_DIM / BM), 256, 0, stream>>>(
      Pt, nullptr, Th, Tl, newW, nullptr, nullptr,
      W, u, v, pre0, post0, D_DIM, D_DIM, S_DIM);
}
